// DeformableConv2D_57226144251968
// MI455X (gfx1250) — matrix operand order repaired and checked by reading, not compiled, not run on hardware
//
#include <hip/hip_runtime.h>
#include <hip/hip_bf16.h>

// ---------------------------------------------------------------------------
// Deformable Conv2D for MI455X (gfx1250, wave32, WMMA)
//   B=8, H=W=128, C=64, 3x3 offsets (K2=9), F=128 output filters.
//   Pipeline: [offset conv (scalar)] -> [fused bilinear-sample + bf16 WMMA GEMM]
//   GEMM: M=131072, K=576, N=128 using v_wmma_f32_16x16x32_bf16.
//   Each WG: 64-pixel M-tile; each wave owns one 16-wide N-tile and keeps
//   4 accumulators so every B fragment is reused 4x from registers.
// ---------------------------------------------------------------------------

typedef __bf16 bf16_t;
typedef __attribute__((ext_vector_type(16))) __bf16 v16bf;
typedef __attribute__((ext_vector_type(8)))  __bf16 v8bf;
typedef __attribute__((ext_vector_type(8)))  float  v8f;

#define B_    8
#define H_    128
#define W_    128
#define C_    64
#define K2_   9
#define KC_   576      // K2_ * C_
#define NOUT_ 128
#define NPIX_ (B_ * H_ * W_)
#define OFFCH_ 18      // 2 * K2_
#define MTILE_ 64      // pixels per workgroup
#define MSUB_  4       // 16-row sub-tiles per wave

// Workspace layout (bytes):
//   [0, 147456)                      : wT  bf16 [NOUT_][KC_]  (transposed 1x1-conv weights)
//   [147456, 147456 + NPIX_*18*4)    : offs fp32 [NPIX_][18]
#define WS_WT_BYTES (KC_ * NOUT_ * 2)

// ---------------------------------------------------------------------------
// Kernel 0: convert + transpose w_out [576][128] f32 -> wT [128][576] bf16
// ---------------------------------------------------------------------------
__global__ void wcvt_kernel(const float* __restrict__ w_out, bf16_t* __restrict__ wT) {
    int idx = blockIdx.x * blockDim.x + threadIdx.x;
    if (idx >= KC_ * NOUT_) return;
    int k = idx / NOUT_;
    int n = idx % NOUT_;
    wT[(size_t)n * KC_ + k] = (bf16_t)w_out[idx];
}

// ---------------------------------------------------------------------------
// Kernel 1: 3x3 offset-predicting conv (SAME, zero pad), 64 -> 18 channels.
// One thread per pixel; ~10K FMAs each (2.7 GFLOP total — negligible).
// ---------------------------------------------------------------------------
__global__ void offs_kernel(const float* __restrict__ x,
                            const float* __restrict__ w_off,
                            const float* __restrict__ b_off,
                            float* __restrict__ offs) {
    int m = blockIdx.x * blockDim.x + threadIdx.x;
    if (m >= NPIX_) return;
    int b  = m / (H_ * W_);
    int hw = m % (H_ * W_);
    int h  = hw / W_;
    int w  = hw % W_;

    float acc[OFFCH_];
#pragma unroll
    for (int oc = 0; oc < OFFCH_; ++oc) acc[oc] = b_off[oc];

    for (int ky = 0; ky < 3; ++ky) {
        int yy = h + ky - 1;
        if (yy < 0 || yy >= H_) continue;
        for (int kx = 0; kx < 3; ++kx) {
            int xx = w + kx - 1;
            if (xx < 0 || xx >= W_) continue;
            const float* xp = x + (((size_t)b * H_ + yy) * W_ + xx) * C_;
            const float* wp = w_off + (size_t)(ky * 3 + kx) * C_ * OFFCH_;
            for (int c = 0; c < C_; ++c) {
                float xv = xp[c];
#pragma unroll
                for (int oc = 0; oc < OFFCH_; ++oc)
                    acc[oc] = fmaf(xv, wp[c * OFFCH_ + oc], acc[oc]);
            }
        }
    }
#pragma unroll
    for (int oc = 0; oc < OFFCH_; ++oc)
        offs[(size_t)m * OFFCH_ + oc] = acc[oc];
}

// ---------------------------------------------------------------------------
// Kernel 2: fused deformable sampling + bf16 WMMA GEMM.
//   Grid: NPIX_/64 workgroups, 256 threads (8 wave32) each.
//   Each WG: 64 pixels (M-tile) x 128 outputs (one 16-wide N-tile per wave,
//   4 M sub-tiles -> 4 accumulators, B fragment reused 4x from registers).
// ---------------------------------------------------------------------------
#define AROW_ (KC_ + 16)   // padded LDS row stride (bf16); 1184 B, 16B aligned

__global__ __launch_bounds__(256) void dcn_gemm_kernel(
        const float*  __restrict__ x,
        const float*  __restrict__ offs,
        const bf16_t* __restrict__ wT,
        const float*  __restrict__ b_out,
        float*        __restrict__ out)
{
    __shared__ bf16_t As[MTILE_][AROW_];   // 64 x 576 sampled A-tile (bf16), ~74 KB

    const int t     = threadIdx.x;
    const int mbase = blockIdx.x * MTILE_;

    // ---- Stage 1: deformable bilinear sampling -> LDS (bf16) -------------
    // thread t handles channel (t&63) for (pixel,tap) pairs (t>>6)+4k.
    // Uniform trip count (144 iters for every thread) — no EXEC divergence.
    {
        const int c = t & 63;
        for (int pair = (t >> 6); pair < MTILE_ * K2_; pair += 4) {
            const int p   = pair / K2_;
            const int tap = pair % K2_;
            const int m   = mbase + p;
            const int b   = m / (H_ * W_);
            const int hw  = m % (H_ * W_);
            const int h   = hw / W_;
            const int w   = hw % W_;

            const float xo = offs[(size_t)m * OFFCH_ + tap * 2 + 0];
            const float yo = offs[(size_t)m * OFFCH_ + tap * 2 + 1];

            float nx = fminf(fmaxf((float)w + xo, 0.0f), (float)(W_ - 1));
            float ny = fminf(fmaxf((float)h + yo, 0.0f), (float)(H_ - 1));

            float x0f = floorf(nx);
            float y0f = floorf(ny);
            float x1f = fminf(x0f + 1.0f, (float)(W_ - 1));
            float y1f = fminf(y0f + 1.0f, (float)(H_ - 1));

            int ix0 = (int)x0f, ix1 = (int)x1f;
            int iy0 = (int)y0f, iy1 = (int)y1f;

            float wa = (x1f - nx) * (y1f - ny);
            float wb = (x1f - nx) * (ny - y0f);
            float wc = (nx - x0f) * (y1f - ny);
            float wd = (nx - x0f) * (ny - y0f);

            const float* xb = x + (size_t)b * H_ * W_ * C_ + c;
            float va = xb[((size_t)iy0 * W_ + ix0) * C_];
            float vb = xb[((size_t)iy1 * W_ + ix0) * C_];
            float vc = xb[((size_t)iy0 * W_ + ix1) * C_];
            float vd = xb[((size_t)iy1 * W_ + ix1) * C_];

            float r = wa * va + wb * vb + wc * vc + wd * vd;
            As[p][tap * C_ + c] = (bf16_t)r;
        }
    }
    __syncthreads();

    // ---- Stage 2: WMMA GEMM (4 x 16x16 D tiles per wave) -----------------
    const int wv   = t >> 5;                 // wave id 0..7 == N-tile
    const int lane = t & 31;
    const int half = lane >> 4;              // 0: lanes 0-15, 1: lanes 16-31
    const int mrow = lane & 15;              // A row (within sub-tile)
    const int ncol = (lane & 15) + wv * 16;  // B/D column this lane supplies

    v8f acc[MSUB_];
#pragma unroll
    for (int mt = 0; mt < MSUB_; ++mt) acc[mt] = (v8f){};

    const bf16_t* wrow = wT + (size_t)ncol * KC_;

    for (int kk = 0; kk < KC_; kk += 32) {
        // B fragment (16-bit B 32x16): lane half selects K block of 16;
        // one contiguous 32 B load from the transposed weight row, reused 4x.
        v16bf bfrag = *(const v16bf*)&wrow[kk + half * 16];

#pragma unroll
        for (int mt = 0; mt < MSUB_; ++mt) {
            // A fragment (16-bit A 16x32, contiguous k order matching B):
            //   VGPR 0..3: K = kk + half*16 + {0..7}
            //   VGPR 4..7: K = kk + half*16 + 8 + {0..7}
            v16bf a;
            {
                const v8bf lo = *(const v8bf*)&As[mt * 16 + mrow][kk + half * 16];
                const v8bf hi = *(const v8bf*)&As[mt * 16 + mrow][kk + half * 16 + 8];
#pragma unroll
                for (int i = 0; i < 8; ++i) { a[i] = lo[i]; a[8 + i] = hi[i]; }
            }
            acc[mt] = __builtin_amdgcn_wmma_f32_16x16x32_bf16(
                          /*neg_a=*/false, a, /*neg_b=*/false, bfrag,
                          /*c_mod=*/(short)0, acc[mt],
                          /*reuse_a=*/false, /*reuse_b=*/false);
        }
    }

    // ---- Epilogue: bias + store (32-bit C/D layout: VGPR v -> M=v or v+8)
    const float bias = b_out[ncol];
#pragma unroll
    for (int mt = 0; mt < MSUB_; ++mt) {
#pragma unroll
        for (int v = 0; v < 8; ++v) {
            int m = mt * 16 + ((half == 0) ? v : (v + 8));
            out[((size_t)(mbase + m)) * NOUT_ + ncol] = acc[mt][v] + bias;
        }
    }
}

// ---------------------------------------------------------------------------
extern "C" void kernel_launch(void* const* d_in, const int* in_sizes, int n_in,
                              void* d_out, int out_size, void* d_ws, size_t ws_size,
                              hipStream_t stream) {
    const float* x     = (const float*)d_in[0];
    const float* w_off = (const float*)d_in[1];
    const float* b_off = (const float*)d_in[2];
    const float* w_out = (const float*)d_in[3];
    const float* b_out = (const float*)d_in[4];
    float*       out   = (float*)d_out;

    bf16_t* wT   = (bf16_t*)d_ws;
    float*  offs = (float*)((char*)d_ws + WS_WT_BYTES);

    (void)in_sizes; (void)n_in; (void)out_size; (void)ws_size;

    wcvt_kernel<<<(KC_ * NOUT_ + 255) / 256, 256, 0, stream>>>(w_out, wT);
    offs_kernel<<<(NPIX_ + 255) / 256, 256, 0, stream>>>(x, w_off, b_off, offs);
    dcn_gemm_kernel<<<NPIX_ / MTILE_, 256, 0, stream>>>(x, offs, wT, b_out, out);
}
